// MultiHeadedAttention_77996606095531
// MI455X (gfx1250) — matrix operand order repaired and checked by reading, not compiled, not run on hardware
//
#include <hip/hip_runtime.h>

// ---------------------------------------------------------------------------
// MI455X / gfx1250 structure-biased multi-head attention.
// Wave32, WMMA f32_16x16x32_f16 for all GEMM-like work,
// TDM (tensor_load_to_lds) double-buffered streaming of the structure tensor.
// ---------------------------------------------------------------------------

typedef __attribute__((ext_vector_type(16))) _Float16 v16h;
typedef __attribute__((ext_vector_type(8)))  _Float16 v8h;
typedef __attribute__((ext_vector_type(8)))  float    v8f;
typedef __attribute__((ext_vector_type(4)))  unsigned int u32x4;
typedef __attribute__((ext_vector_type(8)))  int      i32x8;
typedef __attribute__((ext_vector_type(4)))  int      i32x4;

union V16H { v16h v; v8h h[2]; _Float16 e[16]; };

#define WMMA_F16(a, b, c) \
  __builtin_amdgcn_wmma_f32_16x16x32_f16(false, (a), false, (b), (short)0, (c), false, false)

constexpr int   Bc  = 4;
constexpr int   Lc  = 1024;
constexpr int   Dc  = 1024;
constexpr int   Hc  = 16;
constexpr int   DHc = 64;
constexpr int   SCc = 30;   // struct channels
constexpr int   SPc = 32;   // padded qs channels (30 + const-1 + zero)
constexpr float ALPHAc = 1.0f;
constexpr float BETAc  = 1.0f;

// ---------------- WMMA fragment loaders (wave32 layouts, ISA 7.12.2) -------
__device__ inline v16h load_a_f16(const _Float16* base, int row, int ld, int kk, int lane) {
  const int hi = lane >> 4;
  const _Float16* p = base + (size_t)row * ld + kk + hi * 8;
  V16H u;
  u.h[0] = *(const v8h*)(p);
  u.h[1] = *(const v8h*)(p + 16);
  return u.v;
}
__device__ inline v16h load_a_f32cvt(const float* base, int row, int ld, int kk, int lane) {
  const int hi = lane >> 4;
  const float* p = base + (size_t)row * ld + kk + hi * 8;
  V16H u;
#pragma unroll
  for (int t = 0; t < 8; ++t) {
    u.e[t]     = (_Float16)p[t];
    u.e[t + 8] = (_Float16)p[t + 16];
  }
  return u.v;
}
__device__ inline v16h load_b_f16(const _Float16* base, int k0, int ld, int n0, int lane) {
  const int hi = lane >> 4;
  const _Float16* p = base + (size_t)(k0 + hi * 8) * ld + n0 + (lane & 15);
  V16H u;
#pragma unroll
  for (int t = 0; t < 8; ++t) {
    u.e[t]     = p[(size_t)t * ld];
    u.e[t + 8] = p[(size_t)(t + 16) * ld];
  }
  return u.v;
}
__device__ inline v16h load_b_f32cvt(const float* base, int k0, int ld, int n0, int lane) {
  const int hi = lane >> 4;
  const float* p = base + (size_t)(k0 + hi * 8) * ld + n0 + (lane & 15);
  V16H u;
#pragma unroll
  for (int t = 0; t < 8; ++t) {
    u.e[t]     = (_Float16)p[(size_t)t * ld];
    u.e[t + 8] = (_Float16)p[(size_t)(t + 16) * ld];
  }
  return u.v;
}
__device__ inline v16h load_b_f32cvt_pad(const float* base, int ld, int n0, int kmax, int lane) {
  const int hi = lane >> 4;
  V16H u;
#pragma unroll
  for (int s = 0; s < 16; ++s) {
    const int k = (s < 8) ? (hi * 8 + s) : (16 + hi * 8 + (s - 8));
    u.e[s] = (k < kmax) ? (_Float16)base[(size_t)k * ld + n0 + (lane & 15)] : (_Float16)0.0f;
  }
  return u.v;
}

// ---------------- TDM: 2D tile of structure -> LDS -------------------------
// Tile: 16 rows (q) x 960 f32 (32 k * 30 c), row stride 30720 f32.
__device__ inline void tdm_load_struct_tile(const float* gsrc, unsigned lds_off) {
  const unsigned long long ga = (unsigned long long)(uintptr_t)gsrc;
  u32x4 g0;
  g0[0] = 1u;                                              // count=1, user desc
  g0[1] = lds_off;                                         // LDS byte address
  g0[2] = (unsigned)(ga & 0xffffffffu);                    // global_addr lo
  g0[3] = (unsigned)((ga >> 32) & 0x01ffffffu) | (2u << 30); // addr hi | type=2
  i32x8 g1;
  g1[0] = 0x00020000;                // wg_mask=0, data_size=2 (4B), no flags
  g1[1] = (int)(30720u << 16);       // tensor_dim0[15:0]=30720 in bits 63:48
  g1[2] = (int)(16384u << 16);       // tensor_dim0 hi=0 | tensor_dim1 lo=16384
  g1[3] = (int)(960u  << 16);        // tensor_dim1 hi=0 | tile_dim0=960
  g1[4] = 16;                        // tile_dim1=16, tile_dim2=0
  g1[5] = 30720;                     // tensor_dim0_stride lo32 (f32 units)
  g1[6] = 0;
  g1[7] = 0;
  const i32x4 z4 = {0, 0, 0, 0};               // groups 2/3 unused (2D tile)
  const i32x8 z8 = {0, 0, 0, 0, 0, 0, 0, 0};   // extra group (clang-23 form)
  __builtin_amdgcn_tensor_load_to_lds(g0, g1, z4, z4, z8, 0);
}

// ---------------------------------------------------------------------------
// Kernel 1: fused QKV projection GEMM. grid = (rows/16, 1024/16, 3)
// ---------------------------------------------------------------------------
__global__ __launch_bounds__(32)
void proj_qkv_kernel(const float* Xq, const float* Xk, const float* Xv,
                     const float* Wq, const float* bq,
                     const float* Wk, const float* bk,
                     const float* Wv, const float* bv,
                     _Float16* Qh, _Float16* KhT, _Float16* Vh) {
  const int lane = threadIdx.x;
  const int hi = lane >> 4, ln = lane & 15;
  const int m0 = blockIdx.x * 16;
  const int n0 = blockIdx.y * 16;
  const int z  = blockIdx.z;
  const float* X = (z == 0) ? Xq : (z == 1) ? Xk : Xv;
  const float* W = (z == 0) ? Wq : (z == 1) ? Wk : Wv;
  const float* bias = (z == 0) ? bq : (z == 1) ? bk : bv;

  v8f acc = {};
#pragma unroll 4
  for (int kk = 0; kk < Dc; kk += 32) {
    v16h a = load_a_f32cvt(X, m0 + ln, Dc, kk, lane);
    v16h b = load_b_f32cvt(W, kk, Dc, n0, lane);
    acc = WMMA_F16(a, b, acc);
  }
  const int col = n0 + ln;
  const int h = col >> 6, d = col & 63;
#pragma unroll
  for (int r = 0; r < 8; ++r) {
    const int row = m0 + r + 8 * hi;
    const int bb = row >> 10, seq = row & (Lc - 1);
    float v = acc[r] + bias[col];
    if (z == 0) {
      v *= 0.125f;                        // 1/sqrt(64)
      Qh[(((size_t)(bb * Hc + h) * Lc) + seq) * DHc + d] = (_Float16)v;
    } else if (z == 1) {
      KhT[(((size_t)(bb * Hc + h) * DHc) + d) * Lc + seq] = (_Float16)v;
    } else {
      Vh[(((size_t)(bb * Hc + h) * Lc) + seq) * DHc + d] = (_Float16)v;
    }
  }
}

// ---------------------------------------------------------------------------
// Kernel 2: qs[b,h,q,c] = ALPHA * sum_d Qh*Wsk[c,:]; [30]=ALPHA*q.bsk; [31]=0
// ---------------------------------------------------------------------------
__global__ __launch_bounds__(256)
void qs_kernel(const _Float16* Qh, const float* Wsk, const float* bsk, _Float16* qs) {
  const int gid = blockIdx.x * 256 + threadIdx.x;
  const int c = gid & 31;
  const int row = gid >> 5;                          // (b*H + h)*L + q
  const _Float16* qp = Qh + (size_t)row * DHc;
  float acc = 0.0f;
  if (c < SCc) {
    const float* wp = Wsk + (size_t)c * DHc;
#pragma unroll 8
    for (int d = 0; d < DHc; ++d) acc = fmaf((float)qp[d], wp[d], acc);
  } else if (c == SCc) {
#pragma unroll 8
    for (int d = 0; d < DHc; ++d) acc = fmaf((float)qp[d], bsk[d], acc);
  }
  qs[(size_t)row * SPc + c] = (_Float16)(ALPHAc * acc);
}

// ---------------------------------------------------------------------------
// Kernel 3: attention. WG = (b, 16-query tile) x all 16 heads (wave = head).
// structure streamed by TDM, double-buffered in dynamic LDS, shared by heads.
// Two-pass softmax; P.V / t@Wsv via WMMA.
// Dynamic LDS layout (bytes):
//   [0)           s_S : 2 x 16q x 32k x 30c f32   (2 x 61440)
//   [122880)      s_P : 16 waves x 16q x 32 f16   (16384)
//   [139264)      s_m : 16q x 32k u8              (512)
//   [139776)      s_sum: 16 waves x 16 f32        (1024)   total 140800
// ---------------------------------------------------------------------------
__global__ __launch_bounds__(512)
void attn_kernel(const _Float16* Qh, const _Float16* KhT, const _Float16* Vh,
                 const _Float16* qs, const float* structure,
                 const unsigned char* mask, const unsigned char* kpm,
                 const float* Wsv, const float* bsv,
                 float* ctx, float* top_p, float* sums0) {
  extern __shared__ char smem[];
  float*         s_S   = (float*)smem;                        // [2][16][32][30]
  _Float16*      s_P   = (_Float16*)(smem + 122880);          // [16][16][32]
  unsigned char* s_m   = (unsigned char*)(smem + 139264);     // [16][32]
  float*         s_sum = (float*)(smem + 139776);             // [16][16]
  constexpr int SBUF = 16 * 32 * SCc;                          // f32 elems/buffer

  const int tid = threadIdx.x;
  const int wave = tid >> 5, lane = tid & 31;
  const int hi = lane >> 4, ln = lane & 15;
  const int b = blockIdx.x >> 6;
  const int q0 = (blockIdx.x & 63) << 4;
  const int h = wave;

  const size_t head = (size_t)(b * Hc + h);
  const _Float16* Kbase  = KhT + head * DHc * Lc;       // [64][L]
  const _Float16* Vbase  = Vh  + head * Lc * DHc;       // [L][64]
  const _Float16* qsbase = qs  + (head * Lc + q0) * SPc;
  const int qrowA = (int)(head * Lc) + q0 + ln;
  const float* sbase = structure + ((size_t)b * Lc + q0) * Lc * SCc; // row q0, k0=0

  const v16h aq0 = load_a_f16(Qh, qrowA, DHc, 0,  lane);
  const v16h aq1 = load_a_f16(Qh, qrowA, DHc, 32, lane);

  float mrow[8], srow[8];
#pragma unroll
  for (int r = 0; r < 8; ++r) { mrow[r] = -3.0e38f; srow[r] = 0.0f; }
  v8f acc0 = {}, acc1 = {}, acc2 = {}, acc3 = {};
  float tpart[16];
#pragma unroll
  for (int i = 0; i < 16; ++i) tpart[i] = 0.0f;

  for (int pass = 0; pass < 2; ++pass) {
    __syncthreads();
    if (wave == 0)                                   // prologue: chunk 0 -> buf 0
      tdm_load_struct_tile(sbase, 0u);

    for (int it = 0; it < Lc / 32; ++it) {
      const int k0 = it * 32;
      __syncthreads();   // compute of chunk it-1 done -> safe to refill its buffer
      if (wave == 0) {
        if (it + 1 < Lc / 32) {
          tdm_load_struct_tile(sbase + (size_t)(k0 + 32) * SCc,
                               (unsigned)(((it + 1) & 1) * SBUF * 4));
          __builtin_amdgcn_s_wait_tensorcnt((short)1);   // chunk it has landed
        } else {
          __builtin_amdgcn_s_wait_tensorcnt((short)0);
        }
      }
      {  // mask chunk (small, per-thread)
        const int qq = tid >> 5, kk = tid & 31;
        s_m[qq * 32 + kk] =
            (unsigned char)(mask[((size_t)b * Lc + (q0 + qq)) * Lc + (k0 + kk)]
                            | kpm[(size_t)b * Lc + (k0 + kk)]);
      }
      __syncthreads();   // buf[it&1] + mask ready for everyone
      const float* Sb = s_S + (size_t)(it & 1) * SBUF;

      // ---- scores: two 16x16 tiles via WMMA ----------------------------
      v8f s0 = {}, s1 = {};
      s0 = WMMA_F16(aq0, load_b_f16(Kbase, 0,  Lc, k0,      lane), s0);
      s0 = WMMA_F16(aq1, load_b_f16(Kbase, 32, Lc, k0,      lane), s0);
      s1 = WMMA_F16(aq0, load_b_f16(Kbase, 0,  Lc, k0 + 16, lane), s1);
      s1 = WMMA_F16(aq1, load_b_f16(Kbase, 32, Lc, k0 + 16, lane), s1);

      // ---- + structure bias (30-dot, f32 from LDS) + masking -----------
#pragma unroll
      for (int r = 0; r < 8; ++r) {
        const int q = r + 8 * hi;
        const _Float16* qr = qsbase + q * SPc;       // L2-resident
        const float* S0p = Sb + (q * 32 + ln) * SCc;
        const float* S1p = Sb + (q * 32 + 16 + ln) * SCc;
        float sc0 = (float)qr[SCc];                  // const-1 channel (q.bsk)
        float sc1 = sc0;
#pragma unroll
        for (int c = 0; c < SCc; ++c) {
          const float qc = (float)qr[c];
          sc0 = fmaf(qc, S0p[c], sc0);
          sc1 = fmaf(qc, S1p[c], sc1);
        }
        float e0 = s0[r] + sc0;
        float e1 = s1[r] + sc1;
        if (s_m[q * 32 + ln])      e0 = -1.0e18f;
        if (s_m[q * 32 + 16 + ln]) e1 = -1.0e18f;
        s0[r] = e0; s1[r] = e1;
      }

      if (pass == 0) {
#pragma unroll
        for (int r = 0; r < 8; ++r)
          mrow[r] = fmaxf(mrow[r], fmaxf(s0[r], s1[r]));
      } else {
#pragma unroll
        for (int r = 0; r < 8; ++r) {
          const int q = r + 8 * hi;
          const float p0 = __expf(s0[r] - mrow[r]);
          const float p1 = __expf(s1[r] - mrow[r]);
          srow[r] += p0 + p1;
          s_P[(wave * 16 + q) * 32 + ln]      = (_Float16)p0;
          s_P[(wave * 16 + q) * 32 + 16 + ln] = (_Float16)p1;
          if (h == 0) {   // unnormalized top-attn; rescaled by kernel 4
            float* tp = top_p + (((size_t)b * Lc + (q0 + q)) * Lc) + k0;
            tp[ln] = p0;
            tp[16 + ln] = p1;
          }
        }
        __syncthreads();   // P visible (cross-lane) before re-staging

        // ---- context += P(16x32) @ V(32x64) : 4 WMMAs -------------------
        const v16h ap = load_a_f16(s_P + wave * 16 * 32, ln, 32, 0, lane);
        acc0 = WMMA_F16(ap, load_b_f16(Vbase, k0, DHc, 0,  lane), acc0);
        acc1 = WMMA_F16(ap, load_b_f16(Vbase, k0, DHc, 16, lane), acc1);
        acc2 = WMMA_F16(ap, load_b_f16(Vbase, k0, DHc, 32, lane), acc2);
        acc3 = WMMA_F16(ap, load_b_f16(Vbase, k0, DHc, 48, lane), acc3);

        // ---- t[q,c] += sum_k P[q,k]*S[q,k,c]  (lane: q=ln, c-span of 16)
        {
          const int qq = ln, c0 = hi * 16;
          for (int kk = 0; kk < 32; ++kk) {
            const float pv = (float)s_P[(wave * 16 + qq) * 32 + kk];
            const float* Sp = Sb + (qq * 32 + kk) * SCc;
#pragma unroll
            for (int cc = 0; cc < 16; ++cc)
              if (c0 + cc < SCc)
                tpart[cc] = fmaf(pv, Sp[c0 + cc], tpart[cc]);
          }
        }
      }
    }
    if (pass == 0) {   // reduce row-max across the 16-lane half-wave
#pragma unroll
      for (int r = 0; r < 8; ++r)
        for (int d = 1; d < 16; d <<= 1)
          mrow[r] = fmaxf(mrow[r], __shfl_xor(mrow[r], d, 32));
    }
  }

  // ---- reduce row sums, publish -------------------------------------------
#pragma unroll
  for (int r = 0; r < 8; ++r)
    for (int d = 1; d < 16; d <<= 1)
      srow[r] += __shfl_xor(srow[r], d, 32);
  if (ln == 0) {
#pragma unroll
    for (int r = 0; r < 8; ++r) s_sum[wave * 16 + r + 8 * hi] = srow[r];
    if (h == 0) {
#pragma unroll
      for (int r = 0; r < 8; ++r)
        sums0[(size_t)b * Lc + q0 + r + 8 * hi] = srow[r];
    }
  }
  __syncthreads();

  // ---- normalize t, re-stage into s_P (16q x 32c f16) ---------------------
  {
    const int qq = ln, c0 = hi * 16;
    const float inv = 1.0f / s_sum[wave * 16 + qq];
#pragma unroll
    for (int cc = 0; cc < 16; ++cc)
      s_P[(wave * 16 + qq) * 32 + c0 + cc] =
          (c0 + cc < SCc) ? (_Float16)(tpart[cc] * inv) : (_Float16)0.0f;
  }
  __syncthreads();

  // ---- ctx2 = t(16x32) @ Wsv(30->32 x 64) : 4 WMMAs -----------------------
  const v16h at = load_a_f16(s_P + wave * 16 * 32, ln, 32, 0, lane);
  v8f c20 = {}, c21 = {}, c22 = {}, c23 = {};
  c20 = WMMA_F16(at, load_b_f32cvt_pad(Wsv, DHc, 0,  SCc, lane), c20);
  c21 = WMMA_F16(at, load_b_f32cvt_pad(Wsv, DHc, 16, SCc, lane), c21);
  c22 = WMMA_F16(at, load_b_f32cvt_pad(Wsv, DHc, 32, SCc, lane), c22);
  c23 = WMMA_F16(at, load_b_f32cvt_pad(Wsv, DHc, 48, SCc, lane), c23);

  // ---- final: ctx = acc/sum + BETA*(ctx2 + bsv) ---------------------------
#pragma unroll
  for (int r = 0; r < 8; ++r) {
    const int q = r + 8 * hi;
    const float inv = 1.0f / srow[r];
    float* op = ctx + ((size_t)b * Lc + (q0 + q)) * Dc + h * DHc;
    op[ln]      = acc0[r] * inv + BETAc * (c20[r] + bsv[ln]);
    op[16 + ln] = acc1[r] * inv + BETAc * (c21[r] + bsv[16 + ln]);
    op[32 + ln] = acc2[r] * inv + BETAc * (c22[r] + bsv[32 + ln]);
    op[48 + ln] = acc3[r] * inv + BETAc * (c23[r] + bsv[48 + ln]);
  }
}

// ---------------------------------------------------------------------------
// Kernel 4: normalize top_attn rows by their softmax sums.
// ---------------------------------------------------------------------------
__global__ __launch_bounds__(256)
void norm_top_kernel(float* top_p, const float* sums0) {
  const size_t idx = (size_t)blockIdx.x * 256 + threadIdx.x;   // B*L*L
  top_p[idx] = top_p[idx] / sums0[idx >> 10];
}

// ---------------------------------------------------------------------------
// Kernel 5: output projection  out = ctx @ Wo + bo
// ---------------------------------------------------------------------------
__global__ __launch_bounds__(32)
void out_proj_kernel(const float* ctx, const float* Wo, const float* bo, float* out) {
  const int lane = threadIdx.x;
  const int hi = lane >> 4, ln = lane & 15;
  const int m0 = blockIdx.x * 16;
  const int n0 = blockIdx.y * 16;
  v8f acc = {};
#pragma unroll 4
  for (int kk = 0; kk < Dc; kk += 32) {
    v16h a = load_a_f32cvt(ctx, m0 + ln, Dc, kk, lane);
    v16h b = load_b_f32cvt(Wo, kk, Dc, n0, lane);
    acc = WMMA_F16(a, b, acc);
  }
  const int col = n0 + ln;
#pragma unroll
  for (int r = 0; r < 8; ++r) {
    const int row = m0 + r + 8 * hi;
    out[(size_t)row * Dc + col] = acc[r] + bo[col];
  }
}

// ---------------------------------------------------------------------------
extern "C" void kernel_launch(void* const* d_in, const int* in_sizes, int n_in,
                              void* d_out, int out_size, void* d_ws, size_t ws_size,
                              hipStream_t stream) {
  (void)in_sizes; (void)n_in; (void)out_size; (void)ws_size;
  const float* key   = (const float*)d_in[0];
  const float* value = (const float*)d_in[1];
  const float* query = (const float*)d_in[2];
  const float* structure = (const float*)d_in[3];
  const unsigned char* mask = (const unsigned char*)d_in[4];
  const unsigned char* kpm  = (const unsigned char*)d_in[5];
  const float* Wq  = (const float*)d_in[6];   const float* bq  = (const float*)d_in[7];
  const float* Wk  = (const float*)d_in[8];   const float* bk  = (const float*)d_in[9];
  const float* Wv  = (const float*)d_in[10];  const float* bv  = (const float*)d_in[11];
  const float* Wsk = (const float*)d_in[12];  const float* bsk = (const float*)d_in[13];
  const float* Wsv = (const float*)d_in[14];  const float* bsv = (const float*)d_in[15];
  const float* Wo  = (const float*)d_in[16];  const float* bo  = (const float*)d_in[17];

  const size_t NQ  = (size_t)Bc * Hc * Lc * DHc;
  const size_t NQS = (size_t)Bc * Hc * Lc * SPc;
  char* ws = (char*)d_ws;
  _Float16* Qh  = (_Float16*)ws;                 ws += NQ * sizeof(_Float16);
  _Float16* KhT = (_Float16*)ws;                 ws += NQ * sizeof(_Float16);
  _Float16* Vh  = (_Float16*)ws;                 ws += NQ * sizeof(_Float16);
  _Float16* qsb = (_Float16*)ws;                 ws += NQS * sizeof(_Float16);
  float*    ctx = (float*)ws;                    ws += (size_t)Bc * Lc * Dc * sizeof(float);
  float*    sums0 = (float*)ws;                  ws += (size_t)Bc * Lc * sizeof(float);

  float* out   = (float*)d_out;
  float* top_p = out + (size_t)Bc * Lc * Dc;

  proj_qkv_kernel<<<dim3((Bc * Lc) / 16, Dc / 16, 3), 32, 0, stream>>>(
      query, key, value, Wq, bq, Wk, bk, Wv, bv, Qh, KhT, Vh);

  qs_kernel<<<(Bc * Hc * Lc * SPc) / 256, 256, 0, stream>>>(Qh, Wsk, bsk, qsb);

  const size_t attn_lds = 140800;   // see kernel comment
  attn_kernel<<<Bc * (Lc / 16), 512, attn_lds, stream>>>(
      Qh, KhT, Vh, qsb, structure, mask, kpm, Wsv, bsv, ctx, top_p, sums0);

  norm_top_kernel<<<(Bc * Lc * Lc) / 256, 256, 0, stream>>>(top_p, sums0);

  out_proj_kernel<<<dim3((Bc * Lc) / 16, Dc / 16), 32, 0, stream>>>(ctx, Wo, bo, out);
}
